// GPT2Attention_49435073577418
// MI455X (gfx1250) — matrix operand order repaired and checked by reading, not compiled, not run on hardware
//
#include <hip/hip_runtime.h>

// ---------------------------------------------------------------------------
// GPT-2 causal attention (B=4, S=2048, D=1024, HD=128) for gfx1250 (MI455X).
// All GEMMs run on v_wmma_f32_16x16x32_f16 (f32 accumulate). Intermediates
// (q, k, vT) are f16 in workspace (~6.8 MB, L2-resident). Causal mask is
// analytic -> the 16 MB mask input is never read.
// ---------------------------------------------------------------------------

typedef __attribute__((ext_vector_type(16))) _Float16 v16h;
typedef __attribute__((ext_vector_type(8)))  _Float16 v8h;
typedef __attribute__((ext_vector_type(8)))  float    v8f;
typedef __attribute__((ext_vector_type(4)))  float    v4f;

union V16u { v16h v; v8h h[2]; };

#define WMMA_F16(a, b, c) \
  __builtin_amdgcn_wmma_f32_16x16x32_f16(false, (a), false, (b), (short)0, (c), false, false)

constexpr int Bc  = 4;
constexpr int Sc  = 2048;
constexpr int Dc  = 1024;
constexpr int HDc = 128;

// ---------------------------------------------------------------------------
// Kernel 0: W [1024][128] f32 -> WT [128][1024] f16 (B-fragment friendly:
// per-lane 16 contiguous halves along K).
// ---------------------------------------------------------------------------
__global__ void __launch_bounds__(256) gpt2attn_wt_kernel(
    const float* __restrict__ Wq, const float* __restrict__ Wk,
    const float* __restrict__ Wv,
    _Float16* __restrict__ WTq, _Float16* __restrict__ WTk,
    _Float16* __restrict__ WTv) {
  int idx = blockIdx.x * blockDim.x + threadIdx.x;   // 0 .. 131071
  int k = idx >> 7;        // 0..1023
  int n = idx & 127;       // 0..127
  int src = k * HDc + n;
  int dst = n * Dc + k;
  WTq[dst] = (_Float16)Wq[src];
  WTk[dst] = (_Float16)Wk[src];
  WTv[dst] = (_Float16)Wv[src];
}

// ---------------------------------------------------------------------------
// Kernel 1: QKV projection. Block = 8 waves; wave w computes output columns
// [16w, 16w+16) of q, k, v for one 16-row tile of x (rows over B*S).
// q,k stored row-major f16 [B*S][128]; v stored transposed [B][128][S].
// ---------------------------------------------------------------------------
__global__ void __launch_bounds__(256) gpt2attn_qkv_kernel(
    const float* __restrict__ x,
    const _Float16* __restrict__ WTq, const _Float16* __restrict__ WTk,
    const _Float16* __restrict__ WTv,
    _Float16* __restrict__ qh, _Float16* __restrict__ kh,
    _Float16* __restrict__ vT) {
  const int tid  = threadIdx.x;
  const int wave = tid >> 5;
  const int lane = tid & 31;
  const int l16  = lane & 15;
  const int lh   = lane >> 4;
  const int rt   = blockIdx.x;        // 512 tiles of 16 rows over B*S
  const int n0   = wave * 16;

  const float* xrow = x + (size_t)(rt * 16 + l16) * Dc;
  const _Float16* wq = WTq + (size_t)(n0 + l16) * Dc + 16 * lh;
  const _Float16* wk = WTk + (size_t)(n0 + l16) * Dc + 16 * lh;
  const _Float16* wv = WTv + (size_t)(n0 + l16) * Dc + 16 * lh;

  v8f cq = {};
  v8f ck = {};
  v8f cv = {};

  for (int k0 = 0; k0 < Dc; k0 += 32) {
    // A fragment of x (16x32, f32 -> f16). Lane holds row l16;
    // elems 0..15 -> K = k0+16*lh+{0..15} contiguous.
    V16u a;
    const v4f f0 = *(const v4f*)(xrow + k0 + 16 * lh);
    const v4f f1 = *(const v4f*)(xrow + k0 + 16 * lh + 4);
    const v4f f2 = *(const v4f*)(xrow + k0 + 16 * lh + 8);
    const v4f f3 = *(const v4f*)(xrow + k0 + 16 * lh + 12);
#pragma unroll
    for (int j = 0; j < 4; ++j) {
      a.v[j]      = (_Float16)f0[j];
      a.v[j + 4]  = (_Float16)f1[j];
      a.v[j + 8]  = (_Float16)f2[j];
      a.v[j + 12] = (_Float16)f3[j];
    }
    // B fragments: lane holds col n0+l16; K = k0+16*lh+{0..15} contiguous.
    v16h bq = *(const v16h*)(wq + k0);
    v16h bk = *(const v16h*)(wk + k0);
    v16h bv = *(const v16h*)(wv + k0);
    cq = WMMA_F16(a.v, bq, cq);
    ck = WMMA_F16(a.v, bk, ck);
    cv = WMMA_F16(a.v, bv, cv);
  }

  // C layout: VGPR r, lanes 0-15 -> M=r, lanes 16-31 -> M=r+8; N = l16.
#pragma unroll
  for (int r = 0; r < 8; ++r) {
    const int gr  = rt * 16 + r + 8 * lh;   // global row over B*S
    const int col = n0 + l16;               // head dim
    qh[(size_t)gr * HDc + col] = (_Float16)cq[r];
    kh[(size_t)gr * HDc + col] = (_Float16)ck[r];
    const int b = gr >> 11;                 // S == 2048
    const int s = gr & (Sc - 1);
    vT[((size_t)b * HDc + col) * Sc + s] = (_Float16)cv[r];
  }
}

// ---------------------------------------------------------------------------
// Kernel 2: causal flash attention. Block = 4 waves; wave = 16 queries.
// Online softmax in C-fragment layout; P transposed via LDS into A layout.
// ---------------------------------------------------------------------------
__global__ void __launch_bounds__(128) gpt2attn_flash_kernel(
    const _Float16* __restrict__ qh, const _Float16* __restrict__ kh,
    const _Float16* __restrict__ vT, float* __restrict__ out) {
  __shared__ __align__(32) _Float16 ldsP[4][16 * 32];

  const int tid  = threadIdx.x;
  const int wave = tid >> 5;
  const int lane = tid & 31;
  const int l16  = lane & 15;
  const int lh   = lane >> 4;
  const int b     = blockIdx.x >> 5;                    // 32 blocks / batch
  const int qBase = (blockIdx.x & 31) * 64 + wave * 16; // query tile start
  const float scale = 0.08838834764831845f;             // 1/sqrt(128)

  // Q A-fragments over the 4 hd-chunks of 32.
  v16h aq[4];
  {
    const _Float16* qrow = qh + ((size_t)b * Sc + qBase + l16) * HDc;
#pragma unroll
    for (int hc = 0; hc < 4; ++hc) {
      V16u a;
      a.h[0] = *(const v8h*)(qrow + hc * 32 + 16 * lh);
      a.h[1] = *(const v8h*)(qrow + hc * 32 + 16 * lh + 8);
      aq[hc] = a.v;
    }
  }

  float m[8], l[8];
  v8f co[8];
  v8f zero = {};
#pragma unroll
  for (int r = 0; r < 8; ++r) { m[r] = -1e30f; l[r] = 0.0f; }
#pragma unroll
  for (int n = 0; n < 8; ++n) co[n] = zero;

  const _Float16* kbB = kh + (size_t)b * Sc * HDc;
  const _Float16* vbB = vT + (size_t)b * HDc * Sc;

  // Causal: only key tiles with k0 <= max row in this query tile.
  for (int k0 = 0; k0 < qBase + 16; k0 += 32) {
    // ---- scores: two 16x16 fragments covering keys [k0,k0+32) ----
    v8f s0 = {};
    v8f s1 = {};
    const _Float16* kr0 = kbB + (size_t)(k0 + l16) * HDc + 16 * lh;
    const _Float16* kr1 = kr0 + 16 * HDc;
#pragma unroll
    for (int hc = 0; hc < 4; ++hc) {
      v16h b0 = *(const v16h*)(kr0 + hc * 32);
      v16h b1 = *(const v16h*)(kr1 + hc * 32);
      s0 = WMMA_F16(aq[hc], b0, s0);
      s1 = WMMA_F16(aq[hc], b1, s1);
    }

    // ---- online softmax (per row r; rows split across lane halves) ----
    float alpha[8];
#pragma unroll
    for (int r = 0; r < 8; ++r) {
      const int row  = qBase + r + 8 * lh;
      const int key0 = k0 + l16;
      const int key1 = key0 + 16;
      float x0 = (key0 <= row) ? s0[r] * scale : -1e30f;
      float x1 = (key1 <= row) ? s1[r] * scale : -1e30f;
      float tmax = fmaxf(x0, x1);
#pragma unroll
      for (int off = 1; off < 16; off <<= 1)
        tmax = fmaxf(tmax, __shfl_xor(tmax, off, 32));
      const float mn = fmaxf(m[r], tmax);
      alpha[r] = __expf(m[r] - mn);
      const float p0 = __expf(x0 - mn);
      const float p1 = __expf(x1 - mn);
      float ps = p0 + p1;
#pragma unroll
      for (int off = 1; off < 16; off <<= 1)
        ps += __shfl_xor(ps, off, 32);
      l[r] = l[r] * alpha[r] + ps;
      m[r] = mn;
      // stash P in row-major LDS tile [16 rows][32 keys]
      ldsP[wave][(r + 8 * lh) * 32 + l16]      = (_Float16)p0;
      ldsP[wave][(r + 8 * lh) * 32 + 16 + l16] = (_Float16)p1;
    }
#pragma unroll
    for (int n = 0; n < 8; ++n)
#pragma unroll
      for (int r = 0; r < 8; ++r)
        co[n][r] *= alpha[r];

    // wave-internal LDS RAW fence (different lanes wrote what we read)
    asm volatile("s_wait_dscnt 0" ::: "memory");

    // ---- P@V: A fragment of P from LDS, B fragments from vT ----
    V16u ap;
    ap.h[0] = *(const v8h*)(&ldsP[wave][l16 * 32 + 16 * lh]);
    ap.h[1] = *(const v8h*)(&ldsP[wave][l16 * 32 + 16 * lh + 8]);
#pragma unroll
    for (int n = 0; n < 8; ++n) {
      v16h bv = *(const v16h*)(vbB + (size_t)(n * 16 + l16) * Sc + k0 + 16 * lh);
      co[n] = WMMA_F16(ap.v, bv, co[n]);
    }
  }

  // ---- epilogue: normalize and store fp32 ----
#pragma unroll
  for (int r = 0; r < 8; ++r) {
    const float inv = 1.0f / l[r];
    const int row = qBase + r + 8 * lh;
    float* orow = out + ((size_t)b * Sc + row) * HDc;
#pragma unroll
    for (int n = 0; n < 8; ++n)
      orow[n * 16 + l16] = co[n][r] * inv;
  }
}

// ---------------------------------------------------------------------------
// Launch. Inputs: 0=x f32, 1=mask i32 (unused; causal analytic), 2..4=W f32.
// Workspace: WTq/WTk/WTv (f16), q/k row-major f16, v transposed f16 (~6.8MB).
// ---------------------------------------------------------------------------
extern "C" void kernel_launch(void* const* d_in, const int* in_sizes, int n_in,
                              void* d_out, int out_size, void* d_ws, size_t ws_size,
                              hipStream_t stream) {
  const float* x  = (const float*)d_in[0];
  const float* Wq = (const float*)d_in[2];
  const float* Wk = (const float*)d_in[3];
  const float* Wv = (const float*)d_in[4];
  float* out = (float*)d_out;

  char* ws = (char*)d_ws;
  const size_t wt_bytes  = (size_t)HDc * Dc * sizeof(_Float16);   // 256 KB each
  const size_t qkv_bytes = (size_t)Bc * Sc * HDc * sizeof(_Float16); // 2 MB each
  _Float16* WTq = (_Float16*)(ws);
  _Float16* WTk = (_Float16*)(ws + wt_bytes);
  _Float16* WTv = (_Float16*)(ws + 2 * wt_bytes);
  _Float16* qhp = (_Float16*)(ws + 3 * wt_bytes);
  _Float16* khp = (_Float16*)(ws + 3 * wt_bytes + qkv_bytes);
  _Float16* vTp = (_Float16*)(ws + 3 * wt_bytes + 2 * qkv_bytes);

  gpt2attn_wt_kernel<<<(Dc * HDc) / 256, 256, 0, stream>>>(Wq, Wk, Wv, WTq, WTk, WTv);
  gpt2attn_qkv_kernel<<<(Bc * Sc) / 16, 256, 0, stream>>>(x, WTq, WTk, WTv, qhp, khp, vTp);
  gpt2attn_flash_kernel<<<(Bc * Sc) / 64, 128, 0, stream>>>(qhp, khp, vTp, out);
}
